// GeoTexAvatar_67276367724993
// MI455X (gfx1250) — matrix operand order repaired and checked by reading, not compiled, not run on hardware
//
#include <hip/hip_runtime.h>
#include <hip/hip_bf16.h>

// ---------------------------------------------------------------------------
// GeoTexAvatar forward for MI455X (gfx1250, wave32, WMMA f16->f32)
// ---------------------------------------------------------------------------

typedef _Float16 half_t;
typedef half_t v16h __attribute__((ext_vector_type(16)));
typedef half_t v8h  __attribute__((ext_vector_type(8)));
typedef float  v8f  __attribute__((ext_vector_type(8)));

#define NPTS  8192
#define NVERT 6890
#define NJ    24
#define CHUNK 2048   // verts per LDS tile (24 KB)

// ---------------------------------------------------------------------------
// Invert 24 affine joint matrices (bottom row is [0,0,0,1]).
// ---------------------------------------------------------------------------
__global__ void invert_mats_kernel(const float* __restrict__ mats,  // (24,4,4)
                                   float* __restrict__ out)         // (24,12)
{
    int j = threadIdx.x;
    if (j >= NJ) return;
    const float* m = mats + j * 16;
    float a = m[0], b = m[1], c = m[2];
    float d = m[4], e = m[5], f = m[6];
    float g = m[8], h = m[9], i = m[10];
    float A = e * i - f * h, B = c * h - b * i, C = b * f - c * e;
    float D = f * g - d * i, E = a * i - c * g, F = c * d - a * f;
    float G = d * h - e * g, H = b * g - a * h, I = a * e - b * d;
    float inv = 1.0f / (a * A + b * D + c * G);
    float r00 = A * inv, r01 = B * inv, r02 = C * inv;
    float r10 = D * inv, r11 = E * inv, r12 = F * inv;
    float r20 = G * inv, r21 = H * inv, r22 = I * inv;
    float tx = m[3], ty = m[7], tz = m[11];
    float* o = out + j * 12;
    o[0] = r00; o[1] = r01; o[2]  = r02; o[3]  = -(r00 * tx + r01 * ty + r02 * tz);
    o[4] = r10; o[5] = r11; o[6]  = r12; o[7]  = -(r10 * tx + r11 * ty + r12 * tz);
    o[8] = r20; o[9] = r21; o[10] = r22; o[11] = -(r20 * tx + r21 * ty + r22 * tz);
}

// ---------------------------------------------------------------------------
// Fused: brute-force 1-NN (LDS-tiled verts) -> nn-weight LBS unpose ->
// normalize -> 24ch trilinear weight-volume sample -> refined LBS unpose.
// ---------------------------------------------------------------------------
__global__ void __launch_bounds__(256)
knn_skin_volume_kernel(const float* __restrict__ wpts,    // (N,3)
                       const float* __restrict__ verts,   // (V,3)
                       const float* __restrict__ skinw,   // (V,24)
                       const float* __restrict__ l2c,     // (24,12)
                       const float* __restrict__ bounds,  // (2,3)
                       const float* __restrict__ vol,     // (24,128,128,128)
                       float* __restrict__ cano_pts,      // (N,3)
                       int* __restrict__ nearf)           // (N)
{
    __shared__ float sv[CHUNK * 3];
    const int tid = threadIdx.x;
    const int n = blockIdx.x * blockDim.x + tid;
    const float px = wpts[n * 3 + 0], py = wpts[n * 3 + 1], pz = wpts[n * 3 + 2];

    float best = 3.4e38f;
    int bidx = 0;
    for (int base = 0; base < NVERT; base += CHUNK) {
        int cnt = min(CHUNK, NVERT - base);
        for (int i = tid; i < cnt * 3; i += blockDim.x) sv[i] = verts[base * 3 + i];
        if (base + CHUNK < NVERT)
            __builtin_prefetch(verts + (base + CHUNK) * 3 + tid, 0, 1);
        __syncthreads();
        for (int v = 0; v < cnt; ++v) {
            float dx = px - sv[v * 3 + 0];
            float dy = py - sv[v * 3 + 1];
            float dz = pz - sv[v * 3 + 2];
            float d2 = dx * dx + dy * dy + dz * dz;
            if (d2 < best) { best = d2; bidx = base + v; }
        }
        __syncthreads();
    }
    nearf[n] = (best < 0.08f * 0.08f) ? 1 : 0;

    // First skinning with nearest-vertex weights.
    float M[12];
#pragma unroll
    for (int r = 0; r < 12; ++r) M[r] = 0.0f;
    for (int j = 0; j < NJ; ++j) {
        float wj = skinw[bidx * NJ + j];
#pragma unroll
        for (int r = 0; r < 12; ++r) M[r] += wj * l2c[j * 12 + r];
    }
    float cx = M[0] * px + M[1] * py + M[2]  * pz + M[3];
    float cy = M[4] * px + M[5] * py + M[6]  * pz + M[7];
    float cz = M[8] * px + M[9] * py + M[10] * pz + M[11];

    float mnx = bounds[0], mny = bounds[1], mnz = bounds[2];
    float mxx = bounds[3], mxy = bounds[4], mxz = bounds[5];
    float ux = (cx - mnx) / (mxx - mnx);
    float uy = (cy - mny) / (mxy - mny);
    float uz = (cz - mnz) / (mxz - mnz);

    // Trilinear sample (align_corners=True, border clamp), D=128.
    float gx = fminf(fmaxf(ux * 127.0f, 0.0f), 127.0f);
    float gy = fminf(fmaxf(uy * 127.0f, 0.0f), 127.0f);
    float gz = fminf(fmaxf(uz * 127.0f, 0.0f), 127.0f);
    int x0 = (int)floorf(gx), y0 = (int)floorf(gy), z0 = (int)floorf(gz);
    int x1 = min(x0 + 1, 127), y1 = min(y0 + 1, 127), z1 = min(z0 + 1, 127);
    float tx = gx - (float)x0, ty = gy - (float)y0, tz = gz - (float)z0;
    float w000 = (1 - tx) * (1 - ty) * (1 - tz);
    float w100 = tx * (1 - ty) * (1 - tz);
    float w010 = (1 - tx) * ty * (1 - tz);
    float w001 = (1 - tx) * (1 - ty) * tz;
    float w110 = tx * ty * (1 - tz);
    float w101 = tx * (1 - ty) * tz;
    float w011 = (1 - tx) * ty * tz;
    float w111 = tx * ty * tz;
    size_t o000 = (size_t)x0 * 16384 + (size_t)y0 * 128 + z0;
    size_t o100 = (size_t)x1 * 16384 + (size_t)y0 * 128 + z0;
    size_t o010 = (size_t)x0 * 16384 + (size_t)y1 * 128 + z0;
    size_t o001 = (size_t)x0 * 16384 + (size_t)y0 * 128 + z1;
    size_t o110 = (size_t)x1 * 16384 + (size_t)y1 * 128 + z0;
    size_t o101 = (size_t)x1 * 16384 + (size_t)y0 * 128 + z1;
    size_t o011 = (size_t)x0 * 16384 + (size_t)y1 * 128 + z1;
    size_t o111 = (size_t)x1 * 16384 + (size_t)y1 * 128 + z1;

    float M2[12];
#pragma unroll
    for (int r = 0; r < 12; ++r) M2[r] = 0.0f;
    for (int j = 0; j < NJ; ++j) {
        const float* vj = vol + (size_t)j * 2097152;
        float wj = vj[o000] * w000 + vj[o100] * w100 + vj[o010] * w010 + vj[o001] * w001
                 + vj[o110] * w110 + vj[o101] * w101 + vj[o011] * w011 + vj[o111] * w111;
#pragma unroll
        for (int r = 0; r < 12; ++r) M2[r] += wj * l2c[j * 12 + r];
    }
    cano_pts[n * 3 + 0] = M2[0] * px + M2[1] * py + M2[2]  * pz + M2[3];
    cano_pts[n * 3 + 1] = M2[4] * px + M2[5] * py + M2[6]  * pz + M2[7];
    cano_pts[n * 3 + 2] = M2[8] * px + M2[9] * py + M2[10] * pz + M2[11];
}

// ---------------------------------------------------------------------------
// Pose feature bilinear sample (64ch, 512x512, align_corners, border clamp)
// -> builds f16 warp-field input [cano_pts(3), pose_feat(64)], lda = 72.
// ---------------------------------------------------------------------------
__global__ void __launch_bounds__(256)
pose_feat_kernel(const float* __restrict__ cano_pts,
                 const float* __restrict__ center,   // (3)
                 const float* __restrict__ fmap,     // (64,512,512)
                 half_t* __restrict__ in_feat)       // (N,72)
{
    int n = blockIdx.x * blockDim.x + threadIdx.x;
    float cx = cano_pts[n * 3 + 0];
    float cy = cano_pts[n * 3 + 1];
    float cz = cano_pts[n * 3 + 2];
    half_t* row = in_feat + (size_t)n * 72;
    row[0] = (half_t)cx; row[1] = (half_t)cy; row[2] = (half_t)cz;

    float gx = cx - center[0];
    float gy = -(cy - center[1]);
    float ix = fminf(fmaxf((gx + 1.0f) * 0.5f * 511.0f, 0.0f), 511.0f);
    float iy = fminf(fmaxf((gy + 1.0f) * 0.5f * 511.0f, 0.0f), 511.0f);
    int x0 = (int)floorf(ix), y0 = (int)floorf(iy);
    int x1 = min(x0 + 1, 511), y1 = min(y0 + 1, 511);
    float tx = ix - (float)x0, ty = iy - (float)y0;
    float w00 = (1 - tx) * (1 - ty), w10 = tx * (1 - ty);
    float w01 = (1 - tx) * ty,       w11 = tx * ty;
    size_t o00 = (size_t)y0 * 512 + x0, o10 = (size_t)y0 * 512 + x1;
    size_t o01 = (size_t)y1 * 512 + x0, o11 = (size_t)y1 * 512 + x1;
    for (int c = 0; c < 64; ++c) {
        const float* f = fmap + (size_t)c * 262144;
        float v = f[o00] * w00 + f[o10] * w10 + f[o01] * w01 + f[o11] * w11;
        row[3 + c] = (half_t)v;
    }
}

// ---------------------------------------------------------------------------
// f32 -> f16 weight conversion with padded row stride (zero fill).
// ---------------------------------------------------------------------------
__global__ void conv_w_kernel(const float* __restrict__ src, half_t* __restrict__ dst,
                              int rows, int cols, int ldst)
{
    int i = blockIdx.x * blockDim.x + threadIdx.x;
    if (i >= rows * ldst) return;
    int r = i / ldst, c = i % ldst;
    dst[i] = (c < cols) ? (half_t)src[r * cols + c] : (half_t)0.0f;
}

// ---------------------------------------------------------------------------
// WMMA GEMM:  Out(M,N) = act( A(M,K) * W(N,K)^T + bias )
// A, W, Out in f16; accumulate f32 via v_wmma_f32_16x16x32_f16.
// Each wave computes a 16(M) x 32(N) block: the A fragment is loaded ONCE
// per K-chunk and reused for two B tiles (halves vmem traffic per wmma).
// Block = 4 waves -> 16 x 128 output slab. Grid = (M/16, N/128).
// A/B use contiguous K per lane half (k = 16*hi + e), same order for both; C/D per ISA (wave32).
// ---------------------------------------------------------------------------
__global__ void __launch_bounds__(128)
gemm_wmma_kernel(const half_t* __restrict__ A, int lda,
                 const half_t* __restrict__ W, int ldw,
                 const float* __restrict__ bias,
                 half_t* __restrict__ Out, int ldo,
                 int K, int N, int act)
{
    const int lane = threadIdx.x & 31;
    const int wave = threadIdx.x >> 5;
    const int lo = lane & 15;
    const int hi = lane >> 4;
    const int tileM = blockIdx.x * 16;
    const int tileN = blockIdx.y * 128 + wave * 32;   // two 16-wide sub-tiles

    const half_t* Arow  = A + (size_t)(tileM + lo) * lda;
    const half_t* Wrow0 = W + (size_t)(tileN + lo) * ldw;
    const half_t* Wrow1 = W + (size_t)(tileN + 16 + lo) * ldw;

    // C/D: lane -> column (lo); VGPR r -> row r + hi*8. Bias is per-column.
    float bv0 = bias[tileN + lo];
    float bv1 = bias[tileN + 16 + lo];
    v8f acc0, acc1;
#pragma unroll
    for (int r = 0; r < 8; ++r) { acc0[r] = bv0; acc1[r] = bv1; }

    int kb = 0;
    for (; kb + 32 <= K; kb += 32) {
        // A 16x32 f16: lanes 0-15 row M=lo; elems 0-15 -> K=kb+hi*16..+15
        //              (two aligned v8h loads)
        union { v16h v; v8h h[2]; } ua;
        ua.h[0] = *(const v8h*)(Arow + kb + hi * 16);
        ua.h[1] = *(const v8h*)(Arow + kb + hi * 16 + 8);
        // B 32x16 f16: lane -> column N=lo; elems 0-15 -> K=kb+hi*16..+15
        v16h b0 = *(const v16h*)(Wrow0 + kb + hi * 16);
        v16h b1 = *(const v16h*)(Wrow1 + kb + hi * 16);
        acc0 = __builtin_amdgcn_wmma_f32_16x16x32_f16(false, ua.v, false, b0,
                                                      (short)0, acc0, false, false);
        acc1 = __builtin_amdgcn_wmma_f32_16x16x32_f16(false, ua.v, false, b1,
                                                      (short)0, acc1, false, false);
    }
    if (kb < K) {  // tail chunk with zero padding
        v16h a, b0, b1;
#pragma unroll
        for (int e = 0; e < 16; ++e) {
            int ka = kb + e + hi * 16;
            a[e] = (ka < K) ? Arow[ka] : (half_t)0.0f;
            int kk = kb + e + hi * 16;
            b0[e] = (kk < K) ? Wrow0[kk] : (half_t)0.0f;
            b1[e] = (kk < K) ? Wrow1[kk] : (half_t)0.0f;
        }
        acc0 = __builtin_amdgcn_wmma_f32_16x16x32_f16(false, a, false, b0,
                                                      (short)0, acc0, false, false);
        acc1 = __builtin_amdgcn_wmma_f32_16x16x32_f16(false, a, false, b1,
                                                      (short)0, acc1, false, false);
    }
#pragma unroll
    for (int r = 0; r < 8; ++r) {
        float v0 = acc0[r], v1 = acc1[r];
        if (act == 1)      { v0 = fmaxf(v0, 0.0f); v1 = fmaxf(v1, 0.0f); }
        else if (act == 2) { v0 = (v0 > 0.0f) ? v0 : 0.01f * v0;   // leaky_relu(0.01)
                             v1 = (v1 > 0.0f) ? v1 : 0.01f * v1; }
        size_t rowoff = (size_t)(tileM + r + hi * 8) * ldo;
        Out[rowoff + (tileN + lo)]      = (half_t)v0;
        Out[rowoff + (tileN + 16 + lo)] = (half_t)v1;
    }
}

// ---------------------------------------------------------------------------
// Offset head (256 -> 3): offsets = h*out_w^T + out_b; cano_pts += offsets;
// also writes offsets directly to the output buffer.
// ---------------------------------------------------------------------------
__global__ void __launch_bounds__(256)
offset_out_kernel(const half_t* __restrict__ h,    // (N,256)
                  const float* __restrict__ w3,    // (3,256)
                  const float* __restrict__ b3,    // (3)
                  float* __restrict__ cano_pts,
                  float* __restrict__ out_off)     // (N,3)
{
    int n = blockIdx.x * blockDim.x + threadIdx.x;
    const half_t* hr = h + (size_t)n * 256;
    for (int o = 0; o < 3; ++o) {
        const float* wr = w3 + o * 256;
        float s = b3[o];
        for (int k = 0; k < 256; ++k) s += (float)hr[k] * wr[k];
        out_off[n * 3 + o] = s;
        cano_pts[n * 3 + o] += s;
    }
}

// ---------------------------------------------------------------------------
// Positional encoding: [x, (sin(x*2^f), cos(x*2^f)) for f=0..9] -> 63 dims,
// stored f16 with lda = 64.
// ---------------------------------------------------------------------------
__global__ void __launch_bounds__(256)
posenc_kernel(const float* __restrict__ cano_pts, half_t* __restrict__ pe)
{
    int n = blockIdx.x * blockDim.x + threadIdx.x;
    float x0 = cano_pts[n * 3 + 0];
    float x1 = cano_pts[n * 3 + 1];
    float x2 = cano_pts[n * 3 + 2];
    half_t* row = pe + (size_t)n * 64;
    row[0] = (half_t)x0; row[1] = (half_t)x1; row[2] = (half_t)x2;
    float fr = 1.0f;
    for (int f = 0; f < 10; ++f) {
        float a0 = x0 * fr, a1 = x1 * fr, a2 = x2 * fr;
        row[3 + f * 6 + 0] = (half_t)sinf(a0);
        row[3 + f * 6 + 1] = (half_t)sinf(a1);
        row[3 + f * 6 + 2] = (half_t)sinf(a2);
        row[3 + f * 6 + 3] = (half_t)cosf(a0);
        row[3 + f * 6 + 4] = (half_t)cosf(a1);
        row[3 + f * 6 + 5] = (half_t)cosf(a2);
        fr *= 2.0f;
    }
}

// Skip-concat for shared MLP layer 4: [h(256), pe(63)] -> 319 wide (lda 320).
__global__ void concat_kernel(const half_t* __restrict__ h,   // (N,256)
                              const half_t* __restrict__ pe,  // (N,64) 63 valid
                              half_t* __restrict__ out)       // (N,320)
{
    int i = blockIdx.x * blockDim.x + threadIdx.x;
    if (i >= NPTS * 319) return;
    int n = i / 319, c = i % 319;
    out[(size_t)n * 320 + c] = (c < 256) ? h[(size_t)n * 256 + c]
                                         : pe[(size_t)n * 64 + (c - 256)];
}

// ---------------------------------------------------------------------------
// Final fused head: geo (128->2), rgb (128->3, sigmoid), mask, alpha, occ.
// Output layout: raw (N,4) | occ (N) | offsets (N,3)
// ---------------------------------------------------------------------------
__global__ void __launch_bounds__(256)
final_kernel(const half_t* __restrict__ ghid,   // (N,128)
             const half_t* __restrict__ chid,   // (N,128)
             const float* __restrict__ gw1, const float* __restrict__ gb1,  // (2,128),(2)
             const float* __restrict__ cw2, const float* __restrict__ cb2,  // (3,128),(3)
             const float* __restrict__ cano_pts,
             const float* __restrict__ bounds,
             const int* __restrict__ nearf,
             const float* __restrict__ dists,
             float* __restrict__ out)
{
    int n = blockIdx.x * blockDim.x + threadIdx.x;
    const half_t* g = ghid + (size_t)n * 128;
    float ge[2];
    for (int o = 0; o < 2; ++o) {
        const float* wr = gw1 + o * 128;
        float s = gb1[o];
        for (int k = 0; k < 128; ++k) s += (float)g[k] * wr[k];
        ge[o] = s;
    }
    const half_t* ch = chid + (size_t)n * 128;
    float rgb[3];
    for (int o = 0; o < 3; ++o) {
        const float* wr = cw2 + o * 128;
        float s = cb2[o];
        for (int k = 0; k < 128; ++k) s += (float)ch[k] * wr[k];
        rgb[o] = 1.0f / (1.0f + expf(-s));
    }
    float occ = 1.0f / (1.0f + expf(-ge[0]));
    float alpha = fmaxf(ge[1], 0.0f);
    float cx = cano_pts[n * 3 + 0], cy = cano_pts[n * 3 + 1], cz = cano_pts[n * 3 + 2];
    bool inside = (cx > bounds[0] && cx < bounds[3]) &&
                  (cy > bounds[1] && cy < bounds[4]) &&
                  (cz > bounds[2] && cz < bounds[5]);
    bool mask = (!inside) || (nearf[n] == 0);
    if (mask) alpha = 0.0f;
    alpha = 1.0f - expf(-alpha * dists[n]);
    out[n * 4 + 0] = rgb[0];
    out[n * 4 + 1] = rgb[1];
    out[n * 4 + 2] = rgb[2];
    out[n * 4 + 3] = alpha;
    out[NPTS * 4 + n] = occ;
}

// ---------------------------------------------------------------------------
// Host side
// ---------------------------------------------------------------------------
static inline void* ws_take(void* ws, size_t& off, size_t bytes)
{
    void* p = (char*)ws + off;
    off += (bytes + 255) & ~(size_t)255;
    return p;
}

static void launch_gemm(const half_t* A, int lda, const half_t* W, int ldw,
                        const float* bias, half_t* out, int ldo,
                        int K, int N, int act, hipStream_t s)
{
    dim3 grid(NPTS / 16, N / 128);
    gemm_wmma_kernel<<<grid, 128, 0, s>>>(A, lda, W, ldw, bias, out, ldo, K, N, act);
}

static void launch_conv(const float* src, half_t* dst, int rows, int cols, int ldst,
                        hipStream_t s)
{
    int total = rows * ldst;
    conv_w_kernel<<<(total + 255) / 256, 256, 0, s>>>(src, dst, rows, cols, ldst);
}

extern "C" void kernel_launch(void* const* d_in, const int* in_sizes, int n_in,
                              void* d_out, int out_size, void* d_ws, size_t ws_size,
                              hipStream_t stream)
{
    (void)in_sizes; (void)n_in; (void)out_size; (void)ws_size;
    const float* wpts   = (const float*)d_in[0];
    // d_in[1] = viewdirs (unused by reference)
    const float* dists  = (const float*)d_in[2];
    const float* verts  = (const float*)d_in[3];
    const float* c2l    = (const float*)d_in[4];
    const float* bounds = (const float*)d_in[5];
    const float* center = (const float*)d_in[6];
    const float* vol    = (const float*)d_in[7];
    const float* skinw  = (const float*)d_in[8];
    const float* fmap   = (const float*)d_in[9];
    const float *sw[7], *sb[7];
    for (int i = 0; i < 7; ++i) { sw[i] = (const float*)d_in[10 + 2 * i];
                                  sb[i] = (const float*)d_in[11 + 2 * i]; }
    const float* gw0   = (const float*)d_in[24]; const float* gb0 = (const float*)d_in[25];
    const float* gw1   = (const float*)d_in[26]; const float* gb1 = (const float*)d_in[27];
    const float* cw0   = (const float*)d_in[28]; const float* cb0 = (const float*)d_in[29];
    const float* cw1   = (const float*)d_in[30]; const float* cb1 = (const float*)d_in[31];
    const float* cw2   = (const float*)d_in[32]; const float* cb2 = (const float*)d_in[33];
    const float* ow0   = (const float*)d_in[34]; const float* ob0 = (const float*)d_in[35];
    const float* ow1   = (const float*)d_in[36]; const float* ob1 = (const float*)d_in[37];
    const float* ow2   = (const float*)d_in[38]; const float* ob2 = (const float*)d_in[39];
    const float* out_w = (const float*)d_in[40]; const float* out_b = (const float*)d_in[41];

    // ---- workspace layout ----
    size_t off = 0;
    float*  l2c      = (float*)ws_take(d_ws, off, NJ * 12 * sizeof(float));
    float*  cano_pts = (float*)ws_take(d_ws, off, (size_t)NPTS * 3 * sizeof(float));
    int*    nearf    = (int*)  ws_take(d_ws, off, (size_t)NPTS * sizeof(int));
    half_t* in_feat  = (half_t*)ws_take(d_ws, off, (size_t)NPTS * 72 * sizeof(half_t));
    half_t* pe       = (half_t*)ws_take(d_ws, off, (size_t)NPTS * 64 * sizeof(half_t));
    half_t* bufA     = (half_t*)ws_take(d_ws, off, (size_t)NPTS * 256 * sizeof(half_t));
    half_t* bufB     = (half_t*)ws_take(d_ws, off, (size_t)NPTS * 256 * sizeof(half_t));
    half_t* bufC     = (half_t*)ws_take(d_ws, off, (size_t)NPTS * 320 * sizeof(half_t));
    half_t* w_sw[7];
    int sw_ld[7] = {64, 256, 256, 256, 320, 256, 256};
    int sw_ci[7] = {63, 256, 256, 256, 319, 256, 256};
    for (int i = 0; i < 7; ++i)
        w_sw[i] = (half_t*)ws_take(d_ws, off, (size_t)256 * sw_ld[i] * sizeof(half_t));
    half_t* w_gw0 = (half_t*)ws_take(d_ws, off, (size_t)128 * 256 * sizeof(half_t));
    half_t* w_cw0 = (half_t*)ws_take(d_ws, off, (size_t)256 * 256 * sizeof(half_t));
    half_t* w_cw1 = (half_t*)ws_take(d_ws, off, (size_t)128 * 256 * sizeof(half_t));
    half_t* w_ow0 = (half_t*)ws_take(d_ws, off, (size_t)256 * 80  * sizeof(half_t));
    half_t* w_ow1 = (half_t*)ws_take(d_ws, off, (size_t)256 * 256 * sizeof(half_t));
    half_t* w_ow2 = (half_t*)ws_take(d_ws, off, (size_t)256 * 256 * sizeof(half_t));

    float* out_f   = (float*)d_out;
    float* out_off = out_f + NPTS * 5;   // offsets region of output

    // ---- weight conversions (one-time cost per call, tiny) ----
    for (int i = 0; i < 7; ++i) launch_conv(sw[i], w_sw[i], 256, sw_ci[i], sw_ld[i], stream);
    launch_conv(gw0, w_gw0, 128, 256, 256, stream);
    launch_conv(cw0, w_cw0, 256, 256, 256, stream);
    launch_conv(cw1, w_cw1, 128, 256, 256, stream);
    launch_conv(ow0, w_ow0, 256, 67, 80, stream);
    launch_conv(ow1, w_ow1, 256, 256, 256, stream);
    launch_conv(ow2, w_ow2, 256, 256, 256, stream);

    // ---- geometry / sampling pipeline ----
    invert_mats_kernel<<<1, 32, 0, stream>>>(c2l, l2c);
    knn_skin_volume_kernel<<<NPTS / 256, 256, 0, stream>>>(
        wpts, verts, skinw, l2c, bounds, vol, cano_pts, nearf);
    pose_feat_kernel<<<NPTS / 256, 256, 0, stream>>>(cano_pts, center, fmap, in_feat);

    // ---- offset decoder MLP (WMMA) ----
    launch_gemm(in_feat, 72, w_ow0, 80, ob0, bufA, 256, 67, 256, 1, stream);
    launch_gemm(bufA, 256, w_ow1, 256, ob1, bufB, 256, 256, 256, 1, stream);
    launch_gemm(bufB, 256, w_ow2, 256, ob2, bufA, 256, 256, 256, 1, stream);
    offset_out_kernel<<<NPTS / 256, 256, 0, stream>>>(bufA, out_w, out_b, cano_pts, out_off);

    // ---- canonical template: posenc + shared MLP (WMMA) ----
    posenc_kernel<<<NPTS / 256, 256, 0, stream>>>(cano_pts, pe);
    launch_gemm(pe, 64, w_sw[0], 64, sb[0], bufA, 256, 63, 256, 1, stream);
    launch_gemm(bufA, 256, w_sw[1], 256, sb[1], bufB, 256, 256, 256, 1, stream);
    launch_gemm(bufB, 256, w_sw[2], 256, sb[2], bufA, 256, 256, 256, 1, stream);
    launch_gemm(bufA, 256, w_sw[3], 256, sb[3], bufB, 256, 256, 256, 1, stream);
    concat_kernel<<<(NPTS * 319 + 255) / 256, 256, 0, stream>>>(bufB, pe, bufC);
    launch_gemm(bufC, 320, w_sw[4], 320, sb[4], bufA, 256, 319, 256, 1, stream);
    launch_gemm(bufA, 256, w_sw[5], 256, sb[5], bufB, 256, 256, 256, 1, stream);
    launch_gemm(bufB, 256, w_sw[6], 256, sb[6], bufA, 256, 256, 256, 0, stream);  // sh = bufA

    // ---- heads ----
    launch_gemm(bufA, 256, w_gw0, 256, gb0, bufB, 128, 256, 128, 2, stream);  // geo hidden (leaky)
    launch_gemm(bufA, 256, w_cw0, 256, cb0, bufC, 320, 256, 256, 1, stream);  // clr hidden0
    launch_gemm(bufC, 320, w_cw1, 256, cb1, bufA, 128, 256, 128, 1, stream);  // clr hidden1

    final_kernel<<<NPTS / 256, 256, 0, stream>>>(
        bufB, bufA, gw1, gb1, cw2, cb2, cano_pts, bounds, nearf, dists, out_f);
}
